// GSage_MME_4784593567774
// MI455X (gfx1250) — matrix operand order repaired and checked by reading, not compiled, not run on hardware
//
#include <hip/hip_runtime.h>

typedef __attribute__((ext_vector_type(16))) __bf16          v16bf;
typedef __attribute__((ext_vector_type(8)))  float           v8f;
typedef __attribute__((ext_vector_type(16))) unsigned short  v16us;
typedef __attribute__((ext_vector_type(8)))  unsigned short  v8us;

#define EPI_BN        0
#define EPI_RELU_BF16 1
#define EPI_DEC       2
#define EPI_OUT_F32   3

// A fragment: elems {p[0..7], p[16..23]} (p = row + k0 + abase)
__device__ __forceinline__ v16bf ld_a(const unsigned short* __restrict__ p) {
    v8us lo = *(const v8us*)p;
    v8us hi = *(const v8us*)(p + 16);
    v16us h;
#pragma unroll
    for (int i = 0; i < 8; ++i) { h[i] = lo[i]; h[8 + i] = hi[i]; }
    return __builtin_bit_cast(v16bf, h);
}

// B fragment: elems {p[0..7], p[16..23]} (p = Wrow + k0 + ksel)
__device__ __forceinline__ v16bf ld_b(const unsigned short* __restrict__ p) {
    v8us lo = *(const v8us*)p;
    v8us hi = *(const v8us*)(p + 16);
    v16us h;
#pragma unroll
    for (int i = 0; i < 8; ++i) { h[i] = lo[i]; h[8 + i] = hi[i]; }
    return __builtin_bit_cast(v16bf, h);
}

// Fully-unrolled, ping-pong K-panel: wave computes 32 rows x 64 cols
// (2 row sub-tiles x 4 col tiles). A [rows, lda] bf16 zero-padded (lda mult 32),
// W [Mpad, K] bf16 zero-padded (K mult 32) -> no guards in the hot loop.
template <int K>
__device__ __forceinline__ void mm_panel(const unsigned short* __restrict__ A,
                                         const unsigned short* __restrict__ W,
                                         int lda, long long rowoff0, int colblk,
                                         int lane15, int abase, int ksel,
                                         v8f (&acc)[2][4]) {
    constexpr int S = K / 32;
    const unsigned short* pa0 = A + rowoff0 + abase;
    const unsigned short* pa1 = pa0 + (long long)16 * lda;
    const unsigned short* pb[4];
#pragma unroll
    for (int t = 0; t < 4; ++t)
        pb[t] = W + (long long)(colblk + t * 16 + lane15) * K + ksel;

    v16bf a0[2], a1[2], bb[2][4];
    a0[0] = ld_a(pa0);
    a1[0] = ld_a(pa1);
#pragma unroll
    for (int t = 0; t < 4; ++t) bb[0][t] = ld_b(pb[t]);

#pragma unroll
    for (int s = 0; s < S; ++s) {
        const int cur = s & 1, nxt = cur ^ 1;
        if (s + 1 < S) {  // issue next k-step loads before current WMMAs
            const int kn = (s + 1) * 32;
            a0[nxt] = ld_a(pa0 + kn);
            a1[nxt] = ld_a(pa1 + kn);
#pragma unroll
            for (int t = 0; t < 4; ++t) bb[nxt][t] = ld_b(pb[t] + kn);
        }
#pragma unroll
        for (int t = 0; t < 4; ++t) {
            acc[0][t] = __builtin_amdgcn_wmma_f32_16x16x32_bf16(
                false, a0[cur], false, bb[cur][t], (short)0, acc[0][t], false, false);
            acc[1][t] = __builtin_amdgcn_wmma_f32_16x16x32_bf16(
                false, a1[cur], false, bb[cur][t], (short)0, acc[1][t], false, false);
        }
    }
}

// Block = 128 threads = 4 waves; wave w -> 32-row strip, block covers 128 rows x 64 cols.
template <int K, bool DUAL, int EPI>
__global__ void __launch_bounds__(128)
wmma_gemm_kernel(const unsigned short* __restrict__ A1,
                 const unsigned short* __restrict__ A2,
                 const unsigned short* __restrict__ W1,
                 const unsigned short* __restrict__ W2,
                 int N, int M, int lda, int ldc,
                 const float* __restrict__ p0, const float* __restrict__ p1,
                 const float* __restrict__ p2, const float* __restrict__ p3,
                 void* __restrict__ Out, int add_flag) {
    const int wave = threadIdx.x >> 5;
    const int lane = threadIdx.x & 31;
    const int row0 = (blockIdx.y * 4 + wave) * 32;   // N is a multiple of 32
    if (row0 >= N) return;

    const int lane15 = lane & 15;
    const int hi8    = (lane >> 4) * 8;
    const int abase  = (lane & 16) ? 8 : 0;
    const int ksel   = (lane & 16) ? 8 : 0;
    const int colblk = blockIdx.x * 64;
    const long long rowoff0 = (long long)(row0 + lane15) * lda;

    v8f acc[2][4];
#pragma unroll
    for (int rr = 0; rr < 2; ++rr)
#pragma unroll
        for (int t = 0; t < 4; ++t) acc[rr][t] = (v8f)0.0f;

    mm_panel<K>(A1, W1, lda, rowoff0, colblk, lane15, abase, ksel, acc);
    if (DUAL) mm_panel<K>(A2, W2, lda, rowoff0, colblk, lane15, abase, ksel, acc);

    // Epilogue: acc[rr][t][r] is D[row0 + rr*16 + r + hi8][colblk + t*16 + lane15]
#pragma unroll
    for (int t = 0; t < 4; ++t) {
        const int col = colblk + t * 16 + lane15;
        if (EPI == EPI_BN) {
            float g = 0.f, be = 0.f, mu = 0.f, iv = 0.f;
            if (col < M) { g = p0[col]; be = p1[col]; mu = p2[col]; iv = rsqrtf(p3[col] + 1e-5f); }
            __bf16* O = (__bf16*)Out;
#pragma unroll
            for (int rr = 0; rr < 2; ++rr)
#pragma unroll
                for (int r = 0; r < 8; ++r) {
                    int row = row0 + rr * 16 + r + hi8;
                    float y = (col < M) ? (acc[rr][t][r] - mu) * iv * g + be : 0.f;
                    O[(long long)row * ldc + col] = (__bf16)y;  // zero-pad cols >= M
                }
        } else if (EPI == EPI_RELU_BF16) {
            float b = (col < M) ? p0[col] : 0.f;
            __bf16* O = (__bf16*)Out;
#pragma unroll
            for (int rr = 0; rr < 2; ++rr)
#pragma unroll
                for (int r = 0; r < 8; ++r) {
                    int row = row0 + rr * 16 + r + hi8;
                    float y = (col < M) ? fmaxf(acc[rr][t][r] + b, 0.f) : 0.f;
                    O[(long long)row * ldc + col] = (__bf16)y;
                }
        } else if (EPI == EPI_DEC) {
            if (col < M) {
                float b = p0[col];
                float* O = (float*)Out;
#pragma unroll
                for (int rr = 0; rr < 2; ++rr)
#pragma unroll
                    for (int r = 0; r < 8; ++r) {
                        int row = row0 + rr * 16 + r + hi8;
                        long long idx = (long long)row * ldc + col;
                        float v = 0.5f * (acc[rr][t][r] + b);
                        if (add_flag) v += O[idx];
                        O[idx] = v;
                    }
            }
        } else {  // EPI_OUT_F32
            if (col < M) {
                float b = p0[col];
                float* O = (float*)Out;
#pragma unroll
                for (int rr = 0; rr < 2; ++rr)
#pragma unroll
                    for (int r = 0; r < 8; ++r) {
                        int row = row0 + rr * 16 + r + hi8;
                        O[(long long)row * ldc + col] = acc[rr][t][r] + b;
                    }
            }
        }
    }
}

// fp32 [M,K] -> zero-padded bf16 [Mp,Kp]
__global__ void pack_w_kernel(const float* __restrict__ in, __bf16* __restrict__ out,
                              int M, int K, int Mp, int Kp) {
    long long i = (long long)blockIdx.x * blockDim.x + threadIdx.x;
    long long tot = (long long)Mp * Kp;
    if (i >= tot) return;
    int r = (int)(i / Kp), c = (int)(i % Kp);
    float v = (r < M && c < K) ? in[(long long)r * K + c] : 0.f;
    out[i] = (__bf16)v;
}

__global__ void f32_to_bf16_kernel(const float* __restrict__ in,
                                   __bf16* __restrict__ out, long long n) {
    long long i = (long long)blockIdx.x * blockDim.x + threadIdx.x;
    if (i < n) out[i] = (__bf16)in[i];
}

__global__ void fill_u32_kernel(unsigned* __restrict__ p, unsigned v, long long n) {
    long long i = (long long)blockIdx.x * blockDim.x + threadIdx.x;
    if (i < n) p[i] = v;
}

// monotonic key for floats: key order == float order
__device__ __forceinline__ unsigned fkey(float f) {
    unsigned b = __float_as_uint(f);
    return (b >> 31) ? ~b : (b | 0x80000000u);
}

// one thread per (edge, 8-feature chunk): 16B bf16 load + up to 8 atomic max.
// m is a relu output (>= 0) and empty segments decode to 0, so zero values
// (key == fkey(0.0) == 0x80000000) cannot change the result -> skip the atomic.
__global__ void scatter_max_kernel(const unsigned short* __restrict__ m,
                                   const int* __restrict__ src, const int* __restrict__ dst,
                                   unsigned* __restrict__ aggk, long long nwork) {
    long long gid = (long long)blockIdx.x * blockDim.x + threadIdx.x;
    if (gid >= nwork) return;
    long long e = gid >> 3;
    int c = (int)(gid & 7) * 8;
    int s = src[e], d = dst[e];
    v8us v = *(const v8us*)(m + (long long)s * 64 + c);
    unsigned* base = aggk + (long long)d * 64 + c;
#pragma unroll
    for (int i = 0; i < 8; ++i) {
        float f = __uint_as_float(((unsigned)v[i]) << 16);
        unsigned key = fkey(f);
        if (key > 0x80000000u) atomicMax(base + i, key);
    }
}

__global__ void agg_finalize_kernel(const unsigned* __restrict__ aggk,
                                    __bf16* __restrict__ out, long long n) {
    long long i = (long long)blockIdx.x * blockDim.x + threadIdx.x;
    if (i >= n) return;
    unsigned k = aggk[i];
    float f;
    if (k >> 31)       f = __uint_as_float(k ^ 0x80000000u);
    else if (k == 0u)  f = 0.f;                       // empty segment -> 0 (ref)
    else               f = __uint_as_float(~k);
    out[i] = (__bf16)f;
}

extern "C" void kernel_launch(void* const* d_in, const int* in_sizes, int n_in,
                              void* d_out, int out_size, void* d_ws, size_t ws_size,
                              hipStream_t stream) {
    const int N = in_sizes[0] / 512;   // D0 = 512
    const int E = in_sizes[2];

    const float* x0  = (const float*)d_in[0];
    const float* x1  = (const float*)d_in[1];
    const int*   src = (const int*)d_in[2];
    const int*   dst = (const int*)d_in[3];
    auto F = [&](int i) { return (const float*)d_in[i]; };

    // ---- workspace layout ----
    char* w = (char*)d_ws;
    size_t off = 0;
    auto take = [&](size_t bytes) -> void* {
        void* r = w + off;
        off = (off + bytes + 255) & ~(size_t)255;
        return r;
    };
    unsigned short* xb   = (unsigned short*)take((size_t)N * 512 * 2);  // x0/x1 bf16 (reused)
    unsigned short* e1   = (unsigned short*)take((size_t)N * 512 * 2);  // enc hidden (500->512)
    unsigned short* e2   = (unsigned short*)take((size_t)N * 128 * 2);  // enc latent
    float*          feat = (float*)take((size_t)N * 64 * 4);            // 0.5*(d0+d1)
    unsigned short* hA   = (unsigned short*)take((size_t)N * 64 * 2);
    unsigned short* hB   = (unsigned short*)take((size_t)N * 64 * 2);
    unsigned short* mB   = (unsigned short*)take((size_t)N * 64 * 2);
    unsigned*       aggk = (unsigned*)take((size_t)N * 64 * 4);
    unsigned short* aggb = (unsigned short*)take((size_t)N * 64 * 2);
    // packed bf16 weights [Mp, Kp]
    unsigned short* w1a = (unsigned short*)take((size_t)512 * 512 * 2);  // enc0 W1 [500,512]
    unsigned short* w2a = (unsigned short*)take((size_t)128 * 512 * 2);  // enc0 W2 [128,500]
    unsigned short* wda = (unsigned short*)take((size_t)64 * 128 * 2);   // enc0 Wd
    unsigned short* w1b = (unsigned short*)take((size_t)512 * 256 * 2);  // enc1 W1 [500,256]
    unsigned short* w2b = (unsigned short*)take((size_t)128 * 512 * 2);  // enc1 W2 [128,500]
    unsigned short* wdb = (unsigned short*)take((size_t)64 * 128 * 2);   // enc1 Wd
    unsigned short* sw[9];
    for (int i = 0; i < 9; ++i) sw[i] = (unsigned short*)take((size_t)64 * 64 * 2);

    const int cb = 256;
    auto blocks = [&](long long n) { return (int)((n + cb - 1) / cb); };

    // ---- pack weights to padded bf16 ----
    auto pack = [&](int idx, unsigned short* dst_, int M, int K, int Mp, int Kp) {
        long long tot = (long long)Mp * Kp;
        pack_w_kernel<<<blocks(tot), cb, 0, stream>>>(F(idx), (__bf16*)dst_, M, K, Mp, Kp);
    };
    pack(4,  w1a, 500, 512, 512, 512);
    pack(10, w2a, 128, 500, 128, 512);
    pack(16, wda,  64, 128,  64, 128);
    pack(18, w1b, 500, 256, 512, 256);
    pack(24, w2b, 128, 500, 128, 512);
    pack(30, wdb,  64, 128,  64, 128);
    pack(32, sw[0], 64, 64, 64, 64);  // s0_Wp
    pack(34, sw[1], 64, 64, 64, 64);  // s0_Ws
    pack(35, sw[2], 64, 64, 64, 64);  // s0_Wn
    pack(37, sw[3], 64, 64, 64, 64);  // s1_Wp
    pack(39, sw[4], 64, 64, 64, 64);  // s1_Ws
    pack(40, sw[5], 64, 64, 64, 64);  // s1_Wn
    pack(42, sw[6], 64, 64, 64, 64);  // s2_Wp
    pack(44, sw[7],  8, 64, 64, 64);  // s2_Ws (rows 8..63 zero)
    pack(45, sw[8],  8, 64, 64, 64);  // s2_Wn

    const int gy = (N + 127) / 128;      // 128 rows per block (4 waves x 32 rows)
    const dim3 blk(128);
    const long long nf = (long long)N * 64;
    const int nfb = blocks(nf);

    // ---- encoder 0 (D0=512) ----
    f32_to_bf16_kernel<<<blocks((long long)N * 512), cb, 0, stream>>>(x0, (__bf16*)xb, (long long)N * 512);
    wmma_gemm_kernel<512, false, EPI_BN><<<dim3(8, gy), blk, 0, stream>>>(
        xb, nullptr, w1a, nullptr, N, 500, 512, 512, F(6), F(7), F(8), F(9), e1, 0);
    wmma_gemm_kernel<512, false, EPI_BN><<<dim3(2, gy), blk, 0, stream>>>(
        e1, nullptr, w2a, nullptr, N, 128, 512, 128, F(12), F(13), F(14), F(15), e2, 0);
    wmma_gemm_kernel<128, false, EPI_DEC><<<dim3(1, gy), blk, 0, stream>>>(
        e2, nullptr, wda, nullptr, N, 64, 128, 64, F(17), nullptr, nullptr, nullptr, feat, 0);

    // ---- encoder 1 (D1=256) ----
    f32_to_bf16_kernel<<<blocks((long long)N * 256), cb, 0, stream>>>(x1, (__bf16*)xb, (long long)N * 256);
    wmma_gemm_kernel<256, false, EPI_BN><<<dim3(8, gy), blk, 0, stream>>>(
        xb, nullptr, w1b, nullptr, N, 500, 256, 512, F(20), F(21), F(22), F(23), e1, 0);
    wmma_gemm_kernel<512, false, EPI_BN><<<dim3(2, gy), blk, 0, stream>>>(
        e1, nullptr, w2b, nullptr, N, 128, 512, 128, F(26), F(27), F(28), F(29), e2, 0);
    wmma_gemm_kernel<128, false, EPI_DEC><<<dim3(1, gy), blk, 0, stream>>>(
        e2, nullptr, wdb, nullptr, N, 64, 128, 64, F(31), nullptr, nullptr, nullptr, feat, 1);

    f32_to_bf16_kernel<<<nfb, cb, 0, stream>>>(feat, (__bf16*)hA, nf);

    const long long nwork = (long long)E * 8;
    const int ewb = blocks(nwork);

    // ---- SAGE layers ----
    struct Layer { unsigned short *hin, *hout; int wp, ws, wn; int bp, b; };
    const Layer L[3] = {
        {hA, hB, 0, 1, 2, 33, 36},
        {hB, hA, 3, 4, 5, 38, 41},
        {hA, nullptr, 6, 7, 8, 43, 46},
    };
    for (int l = 0; l < 3; ++l) {
        // m = relu(h @ Wp^T + bp)
        wmma_gemm_kernel<64, false, EPI_RELU_BF16><<<dim3(1, gy), blk, 0, stream>>>(
            L[l].hin, nullptr, sw[L[l].wp], nullptr, N, 64, 64, 64,
            F(L[l].bp), nullptr, nullptr, nullptr, mB, 0);
        fill_u32_kernel<<<nfb, cb, 0, stream>>>(aggk, 0u, nf);
        scatter_max_kernel<<<ewb, cb, 0, stream>>>(mB, src, dst, aggk, nwork);
        agg_finalize_kernel<<<nfb, cb, 0, stream>>>(aggk, (__bf16*)aggb, nf);
        // out = h @ Ws^T + agg @ Wn^T + b  (+relu for l<2)
        if (l < 2) {
            wmma_gemm_kernel<64, true, EPI_RELU_BF16><<<dim3(1, gy), blk, 0, stream>>>(
                L[l].hin, aggb, sw[L[l].ws], sw[L[l].wn], N, 64, 64, 64,
                F(L[l].b), nullptr, nullptr, nullptr, L[l].hout, 0);
        } else {
            wmma_gemm_kernel<64, true, EPI_OUT_F32><<<dim3(1, gy), blk, 0, stream>>>(
                L[l].hin, aggb, sw[L[l].ws], sw[L[l].wn], N, 8, 64, 8,
                F(L[l].b), nullptr, nullptr, nullptr, d_out, 0);
        }
    }

    (void)n_in; (void)out_size; (void)ws_size;
}
